// ResidualEmbedding_14302241096040
// MI455X (gfx1250) — matrix operand order repaired and checked by reading, not compiled, not run on hardware
//
#include <hip/hip_runtime.h>
#include <hip/hip_bf16.h>

typedef __attribute__((ext_vector_type(16))) _Float16 v16h;
typedef __attribute__((ext_vector_type(8)))  _Float16 v8h;
typedef __attribute__((ext_vector_type(8)))  float    v8f;
typedef __attribute__((ext_vector_type(4)))  unsigned int v4u;
typedef __attribute__((ext_vector_type(8)))  int    v8i;
typedef __attribute__((ext_vector_type(4)))  int    v4i;

union V16 { v16h v; v8h h[2]; };

#define NPATCH 8192      // 32 images * 16*16 patches
#define NPIX   256       // 16x16 pixels per patch
#define CIN    3
#define COUT   96
#define DIM    768
#define KFLAT  24576     // COUT * NPIX
#define EPS    1e-5f

// ---- workspace layout (in _Float16 units) ----
static constexpr size_t YA_OFF     = 0;                       // y activations f16 [8192][24576]
static constexpr size_t YA_HALFS   = (size_t)NPATCH * KFLAT;  // 201,326,592
static constexpr size_t W2P_OFF    = YA_OFF + YA_HALFS;       // conv2 B-fragments
static constexpr size_t W2P_FRAGS  = 9 * 3 * 6;               // taps * ksteps * ntiles
static constexpr size_t LINB_OFF   = W2P_OFF + W2P_FRAGS * 512;
static constexpr size_t LINB_FRAGS = 768 * 48;                // ksteps * ntiles

__device__ __forceinline__ float gelu_tanh(float v) {
    const float k0 = 0.7978845608028654f, k1 = 0.044715f;
    return 0.5f * v * (1.0f + tanhf(k0 * (v + k1 * v * v * v)));
}

// ---------------------------------------------------------------------------
// Pack conv2_w [96][96][3][3] f32 -> f16 WMMA B-fragments.
// Fragment layout (this code's contiguous k order, matched by A): lane L, element e -> K = 16*(L>>4)+e,
// N = L&15; lane L owns 16 contiguous halfs at frag*512 + L*16.
// Fragment index = ((tap*3 + kstep)*6 + ntile).
// ---------------------------------------------------------------------------
__global__ void pack_w2_kernel(const float* __restrict__ w2, _Float16* __restrict__ w2p) {
    int tid = blockIdx.x * 256 + threadIdx.x;
    if (tid >= (int)(W2P_FRAGS * 32)) return;
    int lane = tid & 31, frag = tid >> 5;
    int ntile = frag % 6;
    int tmp   = frag / 6;
    int ks    = tmp % 3;
    int tap   = tmp / 3;
    int dy = tap / 3, dx = tap % 3;
    int co = ntile * 16 + (lane & 15);
    int kb = ks * 32 + 16 * (lane >> 4);
    _Float16* dst = w2p + (size_t)frag * 512 + lane * 16;
    for (int e = 0; e < 16; ++e) {
        int ci = kb + e;
        dst[e] = (_Float16)w2[((co * COUT + ci) * 3 + dy) * 3 + dx];
    }
}

// Pack lin_w [24576][768] f32 -> f16 B-fragments: frag = kstep*48 + ntile.
__global__ void pack_lin_kernel(const float* __restrict__ lw, _Float16* __restrict__ lbp) {
    size_t tid = (size_t)blockIdx.x * 256 + threadIdx.x;
    if (tid >= LINB_FRAGS * 32) return;
    int    lane = (int)(tid & 31);
    size_t frag = tid >> 5;
    int ntile = (int)(frag % 48);
    int ks    = (int)(frag / 48);
    int n  = ntile * 16 + (lane & 15);
    int kb = ks * 32 + 16 * (lane >> 4);
    _Float16* dst = lbp + frag * 512 + (size_t)lane * 16;
    for (int e = 0; e < 16; ++e)
        dst[e] = (_Float16)lw[(size_t)(kb + e) * DIM + n];
}

// ---------------------------------------------------------------------------
// Fused per-patch residual block. One 256-thread workgroup (8 wave32) per
// patch. GN1->GELU->conv1->GN2->GELU in LDS, conv2 as 9 tap-shifted WMMA
// GEMMs (M=256 pixels, N=96, K=96 per tap), fused 1x1 skip + biases,
// f16 y written in the flatten order k = cout*256 + pixel.
// ---------------------------------------------------------------------------
__global__ __launch_bounds__(256) void fused_block_kernel(
    const float* __restrict__ x,
    const float* __restrict__ gn1_s, const float* __restrict__ gn1_b,
    const float* __restrict__ w1,    const float* __restrict__ b1,
    const float* __restrict__ gn2_s, const float* __restrict__ gn2_b,
    const float* __restrict__ b2,
    const float* __restrict__ pw,    const float* __restrict__ pb,
    const _Float16* __restrict__ w2p,
    _Float16* __restrict__ yA)
{
    __shared__ __align__(16) float    s_raw[CIN][NPIX];   // raw patch (for skip)
    __shared__ float                  s_h1[CIN][18][18];  // GELU(GN1(x)), zero halo
    __shared__ __align__(16) _Float16 s_h2h[NPIX][COUT];  // conv2 A operand
    __shared__ float                  s_stats[6];

    const int tid = threadIdx.x;
    const int pid = blockIdx.x;
    const int b   = pid >> 8;
    const int rem = pid & 255;
    const int pi  = rem >> 4;      // patch row in image
    const int pj  = rem & 15;      // patch col in image
    const int py  = tid >> 4;      // pixel row in patch (thread = pixel)
    const int px  = tid & 15;

    if (tid < 6) s_stats[tid] = 0.0f;
    for (int idx = tid; idx < CIN * 18 * 18; idx += 256) ((float*)s_h1)[idx] = 0.0f;
    __syncthreads();

    // ---- stage raw patch + GN1 stats (C/groups == 1 -> per-channel) ----
    float rawv[CIN];
    for (int c = 0; c < CIN; ++c) {
        size_t g = ((size_t)(b * CIN + c) * 256 + (size_t)(pi * 16 + py)) * 256
                 + (size_t)(pj * 16 + px);
        float v = x[g];
        rawv[c] = v;
        s_raw[c][tid] = v;
        atomicAdd(&s_stats[c], v);
        atomicAdd(&s_stats[3 + c], v * v);
    }
    __syncthreads();

    // ---- GN1 + GELU into padded h1 ----
    for (int c = 0; c < CIN; ++c) {
        float mean = s_stats[c] * (1.0f / 256.0f);
        float var  = s_stats[3 + c] * (1.0f / 256.0f) - mean * mean;
        float v = (rawv[c] - mean) * rsqrtf(var + EPS) * gn1_s[c] + gn1_b[c];
        s_h1[c][py + 1][px + 1] = gelu_tanh(v);
    }
    __syncthreads();
    if (tid < 6) s_stats[tid] = 0.0f;
    __syncthreads();

    // ---- conv1 (K=27, direct VALU) + GN2 partial stats ----
    {
        float ssum[3] = {0.f, 0.f, 0.f}, ssq[3] = {0.f, 0.f, 0.f};
        for (int co = 0; co < COUT; ++co) {
            float acc = b1[co];
            for (int ci = 0; ci < CIN; ++ci)
                for (int dy = 0; dy < 3; ++dy)
                    for (int dx = 0; dx < 3; ++dx)
                        acc += w1[((co * CIN + ci) * 3 + dy) * 3 + dx]
                             * s_h1[ci][py + dy][px + dx];
            s_h2h[tid][co] = (_Float16)acc;
            int g = co >> 5;                    // 3 groups of 32 channels
            ssum[g] += acc; ssq[g] += acc * acc;
        }
        for (int g = 0; g < 3; ++g) {
            atomicAdd(&s_stats[g], ssum[g]);
            atomicAdd(&s_stats[3 + g], ssq[g]);
        }
    }
    __syncthreads();

    // ---- GN2 + GELU in place (group count = 32*256 = 8192) ----
    for (int co = 0; co < COUT; ++co) {
        int g = co >> 5;
        float mean = s_stats[g] * (1.0f / 8192.0f);
        float var  = s_stats[3 + g] * (1.0f / 8192.0f) - mean * mean;
        float v = (float)s_h2h[tid][co];
        v = (v - mean) * rsqrtf(var + EPS) * gn2_s[co] + gn2_b[co];
        s_h2h[tid][co] = (_Float16)gelu_tanh(v);
    }
    __syncthreads();

    // ---- conv2 via WMMA: 9 shifted GEMMs, K=96 in 3 steps of 32 ----
    const int wv = tid >> 5, lane = tid & 31;
    const int lr = lane & 15, lh = lane >> 4;

    v8f acc[2][6];
    for (int mt = 0; mt < 2; ++mt)
        for (int n = 0; n < 6; ++n)
            for (int r = 0; r < 8; ++r) acc[mt][n][r] = 0.0f;

    for (int t = 0; t < 9; ++t) {
        const int dy = t / 3 - 1, dx = t % 3 - 1;
        for (int ks = 0; ks < 3; ++ks) {
            v16h bf[6];
            const _Float16* bbase =
                w2p + (size_t)((t * 3 + ks) * 6) * 512 + (size_t)lane * 16;
            for (int n = 0; n < 6; ++n) bf[n] = *(const v16h*)(bbase + n * 512);

            V16 af[2];
            for (int mt = 0; mt < 2; ++mt) {
                int m  = (wv * 2 + mt) * 16 + lr;        // pixel (A row)
                int yy = (m >> 4) + dy, xx = (m & 15) + dx;
                bool valid = ((unsigned)yy < 16u) && ((unsigned)xx < 16u);
                int pix = valid ? (yy * 16 + xx) : 0;
                const _Float16* s = &s_h2h[pix][ks * 32];
                v8h lo = *(const v8h*)(s + 16 * lh);      // K = 16h .. 16h+7
                v8h hi = *(const v8h*)(s + 16 * lh + 8);  // K = 16h+8 .. 16h+15
                if (!valid)
                    for (int e = 0; e < 8; ++e) { lo[e] = (_Float16)0.0f; hi[e] = (_Float16)0.0f; }
                af[mt].h[0] = lo; af[mt].h[1] = hi;
            }
            for (int mt = 0; mt < 2; ++mt)
                for (int n = 0; n < 6; ++n)
                    acc[mt][n] = __builtin_amdgcn_wmma_f32_16x16x32_f16(
                        false, af[mt].v, false, bf[n], (short)0, acc[mt][n],
                        false, false);
        }
    }

    // ---- epilogue: + conv2_b + 1x1 skip, store f16 y (k = co*256 + pix) ----
    for (int n = 0; n < 6; ++n) {
        int co = n * 16 + lr;
        float p0 = pw[co * 3 + 0], p1 = pw[co * 3 + 1], p2 = pw[co * 3 + 2];
        float bias = pb[co] + b2[co];
        for (int mt = 0; mt < 2; ++mt) {
            int mbase = (wv * 2 + mt) * 16 + 8 * lh;     // 8 consecutive pixels
            v8h outv;
            for (int r = 0; r < 8; ++r) {
                int pix = mbase + r;
                float v = acc[mt][n][r] + bias
                        + p0 * s_raw[0][pix] + p1 * s_raw[1][pix] + p2 * s_raw[2][pix];
                outv[r] = (_Float16)v;
            }
            *(v8h*)(yA + (size_t)pid * KFLAT + (size_t)co * 256 + mbase) = outv;
        }
    }
}

// ---------------------------------------------------------------------------
// Linear: [8192, 24576] @ [24576, 768]. 64x6 grid of 128x128 output tiles.
// A is staged through LDS in K=128 chunks via the Tensor Data Mover (one
// tensor_load_to_lds per chunk, issued by wave 0, tracked by TENSORcnt);
// B fragments come from the packed f16 buffer (38 MB, L2-resident).
// 32 WMMAs per wave between barrier pairs (4x fewer syncs than K=32 staging).
// ---------------------------------------------------------------------------
__global__ __launch_bounds__(256) void linear_gemm_kernel(
    const _Float16* __restrict__ yA,
    const _Float16* __restrict__ lbp,
    const float* __restrict__ lin_b,
    float* __restrict__ out)
{
    __shared__ __align__(16) _Float16 s_A[128][128];   // 32 KB K-chunk

    const int tid = threadIdx.x;
    const int mb  = blockIdx.x;            // 0..63
    const int nb  = blockIdx.y;            // 0..5
    const int wv  = tid >> 5, lane = tid & 31;
    const int lr  = lane & 15, lh = lane >> 4;
    const int mw  = (wv & 3) * 2;          // local M-tile base (0..7)
    const int nq  = (wv >> 2) * 4;         // local N-tile base (0..7)

    v8f acc[2][4];
    for (int mt = 0; mt < 2; ++mt)
        for (int j = 0; j < 4; ++j)
            for (int r = 0; r < 8; ++r) acc[mt][j][r] = 0.0f;

    // TDM descriptor pieces that are chunk-invariant.
    const unsigned int lds_addr = (unsigned int)(unsigned long long)(uintptr_t)&s_A[0][0];
    const unsigned long long ga_row0 =
        (unsigned long long)(uintptr_t)(yA + (size_t)(mb * 128) * KFLAT);

    for (int kc = 0; kc < 192; ++kc) {
        if (wv == 0) {
            // D# group0: count=1 | lds_addr | 57-bit global tile addr | type=2
            unsigned long long ga = ga_row0 + (unsigned long long)kc * 256ull; // bytes
            v4u g0;
            g0[0] = 1u;                                   // count=1, no gather
            g0[1] = lds_addr;                             // LDS dest (bytes)
            g0[2] = (unsigned int)ga;                     // global_addr[31:0]
            g0[3] = ((unsigned int)(ga >> 32) & 0x01FFFFFFu) | (2u << 30); // addr[56:32] | type=2
            // D# group1: mask=0, data_size=2B, tile 128x128, dim0 stride=KFLAT
            v8i g1;
            g1[0] = 0x00010000;            // workgroup_mask=0, data_size=1 (2 bytes)
            g1[1] = (int)0x60000000;       // atomic_addr=0 | tensor_dim0.lo16 (24576)
            g1[2] = (int)0x20000000;       // tensor_dim0.hi16=0 | tensor_dim1.lo16 (8192)
            g1[3] = (int)0x00800000;       // tensor_dim1.hi16=0 | tile_dim0=128
            g1[4] = 0x00000080;            // tile_dim1=128, tile_dim2=0
            g1[5] = KFLAT;                 // tensor_dim0_stride[31:0]
            g1[6] = 0;
            g1[7] = 0;
            v4i z4 = {0, 0, 0, 0};
#if __clang_major__ >= 23
            v8i z8 = {0, 0, 0, 0, 0, 0, 0, 0};
            __builtin_amdgcn_tensor_load_to_lds(g0, g1, z4, z4, z8, 0);
#else
            __builtin_amdgcn_tensor_load_to_lds(g0, g1, z4, z4, 0);
#endif
            __builtin_amdgcn_s_wait_tensorcnt(0);
        }
        __syncthreads();

        for (int kk = 0; kk < 4; ++kk) {
            v16h bf[4];
            const _Float16* bbase =
                lbp + ((size_t)(kc * 4 + kk) * 48 + (size_t)(nb * 8 + nq)) * 512
                    + (size_t)lane * 16;
            for (int j = 0; j < 4; ++j) bf[j] = *(const v16h*)(bbase + j * 512);

            V16 af[2];
            for (int mt = 0; mt < 2; ++mt) {
                const _Float16* s = &s_A[(mw + mt) * 16 + lr][kk * 32];
                af[mt].h[0] = *(const v8h*)(s + 16 * lh);
                af[mt].h[1] = *(const v8h*)(s + 16 * lh + 8);
            }
            for (int mt = 0; mt < 2; ++mt)
                for (int j = 0; j < 4; ++j)
                    acc[mt][j] = __builtin_amdgcn_wmma_f32_16x16x32_f16(
                        false, af[mt].v, false, bf[j], (short)0, acc[mt][j],
                        false, false);
        }
        __syncthreads();
    }

    for (int j = 0; j < 4; ++j) {
        int n = (nb * 8 + nq + j) * 16 + lr;
        float bias = lin_b[n];
        for (int mt = 0; mt < 2; ++mt) {
            int tokbase = mb * 128 + (mw + mt) * 16 + 8 * lh;
            for (int r = 0; r < 8; ++r)
                out[(size_t)(tokbase + r) * DIM + n] = acc[mt][j][r] + bias;
        }
    }
}

extern "C" void kernel_launch(void* const* d_in, const int* in_sizes, int n_in,
                              void* d_out, int out_size, void* d_ws, size_t ws_size,
                              hipStream_t stream) {
    (void)in_sizes; (void)n_in; (void)out_size; (void)ws_size;

    const float* x      = (const float*)d_in[0];
    const float* gn1_s  = (const float*)d_in[1];
    const float* gn1_b  = (const float*)d_in[2];
    const float* conv1w = (const float*)d_in[3];
    const float* conv1b = (const float*)d_in[4];
    const float* gn2_s  = (const float*)d_in[5];
    const float* gn2_b  = (const float*)d_in[6];
    const float* conv2w = (const float*)d_in[7];
    const float* conv2b = (const float*)d_in[8];
    const float* projw  = (const float*)d_in[9];
    const float* projb  = (const float*)d_in[10];
    const float* linw   = (const float*)d_in[11];
    const float* linb   = (const float*)d_in[12];
    float* out = (float*)d_out;

    _Float16* ws  = (_Float16*)d_ws;
    _Float16* yA  = ws + YA_OFF;
    _Float16* w2p = ws + W2P_OFF;
    _Float16* lbp = ws + LINB_OFF;

    // 1) weight repacks (f32 -> f16 WMMA B-fragment layout)
    {
        int th = (int)(W2P_FRAGS * 32);
        pack_w2_kernel<<<(th + 255) / 256, 256, 0, stream>>>(conv2w, w2p);
    }
    {
        size_t th = LINB_FRAGS * 32;
        pack_lin_kernel<<<(unsigned)((th + 255) / 256), 256, 0, stream>>>(linw, lbp);
    }

    // 2) fused residual block: one WG per patch
    fused_block_kernel<<<NPATCH, 256, 0, stream>>>(
        x, gn1_s, gn1_b, conv1w, conv1b, gn2_s, gn2_b, conv2b, projw, projb,
        w2p, yA);

    // 3) token linear: 64 x 6 tiles of 128x128
    linear_gemm_kernel<<<dim3(64, 6), 256, 0, stream>>>(yA, lbp, linb, out);
}
